// BNNConv2d_75170517614982
// MI455X (gfx1250) — 8-bit matrix operand order repaired and checked by reading, compiled, not run on hardware
//
#include <hip/hip_runtime.h>

typedef int   v16i __attribute__((ext_vector_type(16)));
typedef int   v4i  __attribute__((ext_vector_type(4)));
typedef float v8f  __attribute__((ext_vector_type(8)));

#define B_    32
#define CIN_  64
#define H_    128
#define W_    128
#define COUT_ 64

#define FP8_P1 0x38
#define FP8_M1 0xB8

__device__ __forceinline__ int perm_c(int c) {
    int blk = c >> 3, lo3 = c & 7;
    int pos = (blk & 1) ? (4 + (blk >> 1)) : (blk >> 1);
    return pos * 8 + lo3;
}

__global__ void __launch_bounds__(256) bnn_binarize_x(const float* __restrict__ x,
                                                      unsigned char* __restrict__ xs) {
    long i = (long)blockIdx.x * 256 + threadIdx.x;
    int w = (int)(i & (W_ - 1));
    long t = i >> 7;
    int h = (int)(t & (H_ - 1));
    t >>= 7;
    int c = (int)(t & (CIN_ - 1));
    int b = (int)(t >> 6);
    float v = x[i];
    unsigned char s = (v > 0.0f) ? FP8_P1 : ((v < 0.0f) ? FP8_M1 : 0x00);
    xs[(((long)b * H_ + h) * W_ + w) * CIN_ + perm_c(c)] = s;
}

__global__ void __launch_bounds__(256) bnn_binarize_w(const float* __restrict__ wt,
                                                      int* __restrict__ wf) {
    int tid  = blockIdx.x * 256 + threadIdx.x;
    int v    = tid & 7;
    int lane = (tid >> 3) & 31;
    int nt   = (tid >> 8) & 3;
    int t    = tid >> 10;
    int pair = t >> 1, half = t & 1;
    int idx  = ((pair * 4 + nt) * 32 + lane) * 16 + half * 8 + v;
    if (t >= 9) { wf[idx] = 0; return; }
    int kh   = t / 3, kw = t % 3;
    int hiK  = (lane & 16) ? 16 : 0;
    int N    = nt * 16 + (lane & 15);
    unsigned d = 0;
#pragma unroll
    for (int byte = 0; byte < 4; ++byte) {
        int K = (((v * 4 + byte) >> 3) * 16) + (hiK >> 1) + ((v * 4 + byte) & 7);
        float val = wt[((N * CIN_ + K) * 3 + kh) * 3 + kw];
        unsigned char s = (val > 0.0f) ? FP8_P1 : ((val < 0.0f) ? FP8_M1 : 0x00);
        d |= ((unsigned)s) << (8 * byte);
    }
    wf[idx] = (int)d;
}

__global__ void __launch_bounds__(256) bnn_conv_wmma(const unsigned char* __restrict__ xs,
                                                     const int* __restrict__ wf,
                                                     float* __restrict__ out) {
    __shared__ int wlds[10240];
    for (int i = threadIdx.x * 4; i < 10240; i += 1024)
        *(v4i*)&wlds[i] = *(const v4i*)&wf[i];
    __syncthreads();

    const int lane  = threadIdx.x & 31;
    const int wave  = threadIdx.x >> 5;
    const int tile  = blockIdx.x * 8 + wave;
    const long pix0 = (long)tile * 16;
    const int  b    = (int)(pix0 >> 14);
    const int  hp   = (int)(pix0 & 16383);
    const int  h    = hp >> 7;
    const int  wbase = hp & 127;
    const int  m    = lane & 15;
    const int  hi   = lane >> 4;
    const int  hOff = hi * 32;
    const int  lane16 = lane * 16;

    v8f acc[4];
#pragma unroll
    for (int nt = 0; nt < 4; ++nt) acc[nt] = (v8f){};

#pragma unroll
    for (int p = 0; p < 5; ++p) {
        v16i a = (v16i){};
#pragma unroll
        for (int half = 0; half < 2; ++half) {
            const int tap = 2 * p + half;
            if (tap >= 9) break;
            const int kh = tap / 3, kw = tap % 3;
            const int ih = h + kh - 1;
            const int iw = wbase + m + kw - 1;
            const bool ok = ((unsigned)ih < (unsigned)H_) &&
                            ((unsigned)iw < (unsigned)W_);
            if (tap + 2 < 9) {
                const int nih = h + (tap + 2) / 3 - 1;
                __builtin_prefetch(xs + (((long)b * H_ + nih) * W_ + wbase) * CIN_, 0, 3);
            }
            if (ok) {
                const unsigned char* ptr =
                    xs + (((long)b * H_ + ih) * W_ + iw) * CIN_ + hOff;
                v4i q0 = *(const v4i*)(ptr);
                v4i q1 = *(const v4i*)(ptr + 16);
                const int o = 8 * half;
                a[o + 0] = q0.x; a[o + 1] = q0.y; a[o + 2] = q0.z; a[o + 3] = q0.w;
                a[o + 4] = q1.x; a[o + 5] = q1.y; a[o + 6] = q1.z; a[o + 7] = q1.w;
            }
        }
#pragma unroll
        for (int nt = 0; nt < 4; ++nt) {
            const int base = (p * 4 + nt) * 512 + lane16;
            v4i b0 = *(const v4i*)&wlds[base];
            v4i b1 = *(const v4i*)&wlds[base + 4];
            v4i b2 = *(const v4i*)&wlds[base + 8];
            v4i b3 = *(const v4i*)&wlds[base + 12];
            v16i bb;
            bb[0]  = b0.x; bb[1]  = b0.y; bb[2]  = b0.z; bb[3]  = b0.w;
            bb[4]  = b1.x; bb[5]  = b1.y; bb[6]  = b1.z; bb[7]  = b1.w;
            bb[8]  = b2.x; bb[9]  = b2.y; bb[10] = b2.z; bb[11] = b2.w;
            bb[12] = b3.x; bb[13] = b3.y; bb[14] = b3.z; bb[15] = b3.w;
            acc[nt] = __builtin_amdgcn_wmma_f32_16x16x128_fp8_fp8(
                a, bb, (short)0, acc[nt],  false,  false);
        }
    }

#pragma unroll
    for (int nt = 0; nt < 4; ++nt) {
        const int cout = nt * 16 + m;
        float* op = out + (((long)b * COUT_ + cout) * H_ + h) * W_ + wbase + hi * 8;
        float4 s0 = { acc[nt][0], acc[nt][1], acc[nt][2], acc[nt][3] };
        float4 s1 = { acc[nt][4], acc[nt][5], acc[nt][6], acc[nt][7] };
        *(float4*)(op)     = s0;
        *(float4*)(op + 4) = s1;
    }
}

extern "C" void kernel_launch(void* const* d_in, const int* in_sizes, int n_in,
                              void* d_out, int out_size, void* d_ws, size_t ws_size,
                              hipStream_t stream) {
    const float* x  = (const float*)d_in[0];
    const float* wt = (const float*)d_in[1];
    float* y        = (float*)d_out;

    unsigned char* xs = (unsigned char*)d_ws;
    int* wf = (int*)((char*)d_ws + (size_t)33554432);

    bnn_binarize_x<<<131072, 256, 0, stream>>>(x, xs);

    bnn_binarize_w<<<40, 256, 0, stream>>>(wt, wf);

    bnn_conv_wmma<<<4096, 256, 0, stream>>>(xs, wf, y);
}
